// VoiceAutoEncoder_69243462746233
// MI455X (gfx1250) — compile-verified, matrix operand order checked by reading, not run on hardware
//
#include <hip/hip_runtime.h>
#include <hip/hip_bf16.h>

typedef __attribute__((ext_vector_type(16))) _Float16 v16h;
typedef __attribute__((ext_vector_type(8)))  _Float16 v8h;
typedef __attribute__((ext_vector_type(8)))  float    v8f;

#define SRATE   16000
#define NFFT    1024
#define HOP     256
#define NMELS   80
#define GRUH    256
#define ZDIM    16
#define NBINS   65
#define HID     512
#define BATCH   8
#define TLEN    160000
#define PADN    512
#define IRLEN   8000
#define FN      626                 // (TLEN - NFFT)/HOP + 1 with 2*PAD
#define XPN     (TLEN + 2*PADN)    // 161024
#define NFRM    (BATCH*FN)         // 5008 (= 16*313, already M-tile aligned)
#define NBINS_F 513                // NFFT/2 + 1
#define NSMP    (BATCH*TLEN)       // 1280000
#define SCHUNK  1024
#define NBLK    157                // ceil(TLEN/SCHUNK)

#if __has_builtin(__builtin_amdgcn_global_load_async_to_lds_b32) && \
    __has_builtin(__builtin_amdgcn_s_wait_asynccnt)
#define HAVE_ASYNC_LDS 1
#endif

// Async global->LDS copy (ASYNCcnt path, CDNA5). Pointers built via integers:
// global VA for AS1; low 32 bits of a generic LDS pointer are the LDS byte offset.
// Builtin expects non-const AS(1) int* / AS(3) int* parameters.
__device__ __forceinline__ void async_copy_b32(const void* g, void* l) {
#ifdef HAVE_ASYNC_LDS
    typedef __attribute__((address_space(1))) int g_i32;
    typedef __attribute__((address_space(3))) int l_i32;
    __builtin_amdgcn_global_load_async_to_lds_b32(
        (g_i32*)(unsigned long long)(size_t)g,
        (l_i32*)(unsigned long long)(unsigned)(size_t)l, 0, 0);
#else
    *(float*)l = *(const float*)g;
#endif
}
__device__ __forceinline__ void async_wait_all() {
#ifdef HAVE_ASYNC_LDS
    __builtin_amdgcn_s_wait_asynccnt(0);
#endif
}

__device__ __forceinline__ float winf(int i) {
    return 0.5f - 0.5f * cosf(6.2831853072f * (float)i / 1024.0f);
}
__device__ __forceinline__ float sigm(float x) { return 1.0f / (1.0f + __expf(-x)); }
__device__ __forceinline__ int bitrev10(int i) { return (int)(__brev((unsigned)i) >> 22); }
__device__ __forceinline__ unsigned pcg(unsigned v) {
    v = v * 747796405u + 2891336453u;
    unsigned w = ((v >> ((v >> 28u) + 4u)) ^ v) * 277803737u;
    return (w >> 22u) ^ w;
}

// Load one 16x32 f16 WMMA fragment row (per-lane): two contiguous 16B runs -> 2x b128.
// CDNA5 16-bit A/B lane layout: lanes 0-15 hold K={0..7,16..23}, lanes 16-31 K={8..15,24..31}.
__device__ __forceinline__ v16h load_frag(const _Float16* __restrict__ row, int k0, int half) {
    v8h lo = *(const v8h*)(row + k0 + half * 8);
    v8h hi = *(const v8h*)(row + k0 + 16 + half * 8);
    v16h r;
#pragma unroll
    for (int i = 0; i < 8; ++i) { r[i] = lo[i]; r[8 + i] = hi[i]; }
    return r;
}
__device__ __forceinline__ v16h load_frag_lds(const _Float16* row, int k0, int half) {
    v8h lo = *(const v8h*)(row + k0 + half * 8);
    v8h hi = *(const v8h*)(row + k0 + 16 + half * 8);
    v16h r;
#pragma unroll
    for (int i = 0; i < 8; ++i) { r[i] = lo[i]; r[8 + i] = hi[i]; }
    return r;
}

// ---------------- constants: A-weighting, mel filterbank (transposed), wsq inverse ----
__global__ void k_consts(float* aw, float* melfbt, float* wsqi) {
    int idx = blockIdx.x * blockDim.x + threadIdx.x;
    if (idx < NBINS_F) {
        float fr = (float)idx * (8000.0f / 512.0f);
        float fsq = fr * fr;
        float num = 12194.217f * 12194.217f * fsq * fsq;
        float den = (fsq + 20.6f*20.6f) * (fsq + 107.7f*107.7f) * (fsq + 737.9f*737.9f)
                    * sqrtf(fsq + 12194.217f*12194.217f) + 1e-8f;
        aw[idx] = num / den;
    }
    if (idx < NMELS * NBINS_F) {
        int m = idx / NBINS_F, k = idx % NBINS_F;
        float mel_max = 2595.0f * log10f(1.0f + 8000.0f / 700.0f);
        auto fpt = [&](int i) {
            float mm = mel_max * (float)i / 81.0f;
            return 700.0f * (__expf(mm * (2.302585093f / 2595.0f)) - 1.0f);
        };
        float fp0 = fpt(m), fp1 = fpt(m + 1), fp2 = fpt(m + 2);
        float fr = (float)k * (8000.0f / 512.0f);
        float rise = (fr - fp0) / (fp1 - fp0);
        float fall = (fp2 - fr) / (fp2 - fp1);
        melfbt[m * NBINS_F + k] = fmaxf(0.0f, fminf(rise, fall));
    }
    if (idx < XPN) {
        int nlo = (idx - 1023 + 255) / 256; if (nlo < 0) nlo = 0;
        int nhi = idx / 256; if (nhi > FN - 1) nhi = FN - 1;
        float s = 0.0f;
        for (int n = nlo; n <= nhi; ++n) { float w = winf(idx - 256 * n); s += w * w; }
        wsqi[idx] = (s > 1e-11f) ? (1.0f / s) : 1.0f;
    }
}

// ---------------- pack f32 [R,C] -> zero-padded f16 [Rp,Cp] ---------------------------
__global__ void k_pack_f16(const float* __restrict__ src, _Float16* __restrict__ dst,
                           int R, int C, int Rp, int Cp) {
    int idx = blockIdx.x * blockDim.x + threadIdx.x;
    if (idx >= Rp * Cp) return;
    int r = idx / Cp, c = idx % Cp;
    float v = (r < R && c < C) ? src[(size_t)r * C + c] : 0.0f;
    dst[idx] = (_Float16)v;
}

// ---------------- reflect pad:  x[B,TLEN] -> xp[B,XPN] --------------------------------
__global__ void k_pad_reflect(const float* __restrict__ x, float* __restrict__ xp) {
    int idx = blockIdx.x * blockDim.x + threadIdx.x;
    if (idx >= BATCH * XPN) return;
    int b = idx / XPN, i = idx % XPN;
    int j = i - PADN;
    if (j < 0) j = -j;
    if (j >= TLEN) j = 2 * TLEN - 2 - j;
    xp[idx] = x[(size_t)b * TLEN + j];
}

// ---------------- direct autocorrelation F0 (lags 16..319) ----------------------------
__global__ void k_autocorr_f0(const float* __restrict__ xp, float* __restrict__ f0) {
    __shared__ float fx[1024];
    __shared__ float red[320];
    int f = blockIdx.x, b = f / FN, t = f % FN;
    const float* src = xp + (size_t)b * XPN + (size_t)t * HOP;
    int tid = threadIdx.x;
    for (int i = tid; i < 1024; i += 320) async_copy_b32(&src[i], &fx[i]);
    async_wait_all();
    __syncthreads();
    float ss = 0.0f;
    for (int i = tid; i < 1024; i += 320) ss += fx[i] * fx[i];
    red[tid] = ss; __syncthreads();
    if (tid == 0) { float a = 0; for (int i = 0; i < 320; ++i) a += red[i]; red[0] = a; }
    __syncthreads();
    float ac0 = red[0];
    __syncthreads();
    float lagv = -1e30f;
    if (tid < 304) {
        int lag = tid + 16; float s = 0.0f;
        for (int u = 0; u + lag < 1024; ++u) s += fx[u] * fx[u + lag];
        lagv = s;
    }
    red[tid] = lagv; __syncthreads();
    if (tid == 0) {
        float best = -1e30f; int bi = 0;
        for (int i = 0; i < 304; ++i) if (red[i] > best) { best = red[i]; bi = i; }
        float mv = best / (ac0 + 1e-8f);
        f0[f] = (mv < 0.3f) ? 0.0f : (16000.0f / ((float)bi + 16.0f + 1e-8f));
    }
}

// ---------------- 1024-point LDS FFT --------------------------------------------------
__device__ void fft1024_lds(float* re, float* im, int tid, int nthr, bool inverse) {
    for (int s = 1; s <= 10; ++s) {
        int half = 1 << (s - 1);
        float ang0 = (inverse ? 6.2831853072f : -6.2831853072f) / (float)(1 << s);
        for (int j = tid; j < 512; j += nthr) {
            int k = j & (half - 1);
            int i0 = ((j >> (s - 1)) << s) + k;
            int i1 = i0 + half;
            float c = cosf(ang0 * (float)k), sn = sinf(ang0 * (float)k);
            float xr = re[i1], xi = im[i1];
            float tr = xr * c - xi * sn;
            float ti = xr * sn + xi * c;
            float ur = re[i0], ui = im[i0];
            re[i0] = ur + tr; im[i0] = ui + ti;
            re[i1] = ur - tr; im[i1] = ui - ti;
        }
        __syncthreads();
    }
    if (inverse) {
        for (int i = tid; i < 1024; i += nthr) { re[i] *= (1.0f/1024.0f); im[i] *= (1.0f/1024.0f); }
        __syncthreads();
    }
}

// ---------------- windowed STFT: mode 0 -> power spectrum, 1 -> complex ---------------
// Frame bytes go global->LDS via async copies straight into bit-reversed slots
// (per-lane LDS addressing); windowing applied in-place after the ASYNCcnt wait.
__global__ void k_stft(const float* __restrict__ xp, float* __restrict__ specOut,
                       float* __restrict__ reOut, float* __restrict__ imOut, int mode) {
    __shared__ float re[1024], im[1024];
    int f = blockIdx.x, b = f / FN, t = f % FN, tid = threadIdx.x;
    const float* src = xp + (size_t)b * XPN + (size_t)t * HOP;
    for (int i = tid; i < 1024; i += 256) {
        async_copy_b32(&src[i], &re[bitrev10(i)]);
        im[i] = 0.0f;
    }
    async_wait_all();
    __syncthreads();
    for (int i = tid; i < 1024; i += 256) re[bitrev10(i)] *= winf(i);
    __syncthreads();
    fft1024_lds(re, im, tid, 256, false);
    if (mode == 0) {
        for (int k = tid; k <= 512; k += 256)
            specOut[(size_t)f * NBINS_F + k] = re[k] * re[k] + im[k] * im[k];
    } else {
        for (int k = tid; k <= 512; k += 256) {
            reOut[(size_t)f * NBINS_F + k] = re[k];
            imOut[(size_t)f * NBINS_F + k] = im[k];
        }
    }
}

// ---------------- A-weighted loudness -------------------------------------------------
__global__ void k_loud(const float* __restrict__ spec, const float* __restrict__ aw,
                       float* __restrict__ loud) {
    __shared__ float s[256];
    int f = blockIdx.x, tid = threadIdx.x;
    float a = 0.0f;
    for (int k = tid; k < NBINS_F; k += 256)
        a += (spec[(size_t)f * NBINS_F + k] + 1e-8f) * aw[k];
    s[tid] = a; __syncthreads();
    for (int off = 128; off > 0; off >>= 1) { if (tid < off) s[tid] += s[tid + off]; __syncthreads(); }
    if (tid == 0) loud[f] = 10.0f * log10f(s[0] / (float)NBINS_F);
}

// ---------------- mel projection + log ------------------------------------------------
__global__ void k_mel(const float* __restrict__ spec, const float* __restrict__ melfbt,
                      float* __restrict__ mels) {
    __shared__ float sp[NBINS_F];
    int f = blockIdx.x, tid = threadIdx.x;
    for (int k = tid; k < NBINS_F; k += 128) sp[k] = spec[(size_t)f * NBINS_F + k];
    __syncthreads();
    if (tid < NMELS) {
        float a = 0.0f;
        const float* fb = melfbt + (size_t)tid * NBINS_F;
        for (int k = 0; k < NBINS_F; ++k) a += sp[k] * fb[k];
        mels[(size_t)f * NMELS + tid] = logf(a + 1e-5f);
    }
}

// ---------------- per-(batch,mel) normalization over time -----------------------------
__global__ void k_melnorm(float* __restrict__ mels) {
    __shared__ float s1[256], s2[256];
    int b = blockIdx.x / NMELS, m = blockIdx.x % NMELS, tid = threadIdx.x;
    float a = 0.0f, q = 0.0f;
    for (int t = tid; t < FN; t += 256) {
        float v = mels[((size_t)(b * FN + t)) * NMELS + m];
        a += v; q += v * v;
    }
    s1[tid] = a; s2[tid] = q; __syncthreads();
    for (int off = 128; off > 0; off >>= 1) {
        if (tid < off) { s1[tid] += s1[tid + off]; s2[tid] += s2[tid + off]; }
        __syncthreads();
    }
    float mean = s1[0] / (float)FN;
    float var  = s2[0] / (float)FN - mean * mean;
    float inv  = rsqrtf(var + 1e-5f);
    for (int t = tid; t < FN; t += 256) {
        size_t ix = ((size_t)(b * FN + t)) * NMELS + m;
        mels[ix] = (mels[ix] - mean) * inv;
    }
}

// ---------------- WMMA GEMM on packed f16: C[M,N] = A[Mp,KP] @ W[Np,KP]^T + bias ------
// A,W zero-padded f16 (rows multiple of 16, KP multiple of 32, compile-time).
// One wave per 16x16 tile; fully unrolled branch-free K loop of b128 loads + wmma.
template <int KP, int ACT>
__global__ void k_gemm_wmma(const _Float16* __restrict__ A, const _Float16* __restrict__ W,
                            const float* __restrict__ bias, float* __restrict__ C,
                            int ldc, int M, int N) {
    int lane = threadIdx.x & 31;
    int half = lane >> 4;
    int r    = lane & 15;
    int n0 = blockIdx.x * 16;
    int m0 = blockIdx.y * 16;
    const _Float16* Arow = A + (size_t)(m0 + r) * KP;
    const _Float16* Wrow = W + (size_t)(n0 + r) * KP;
    v8f acc = {0.f, 0.f, 0.f, 0.f, 0.f, 0.f, 0.f, 0.f};
#pragma unroll
    for (int k0 = 0; k0 < KP; k0 += 32) {
        v16h a0 = load_frag(Arow, k0, half);
        v16h b0 = load_frag(Wrow, k0, half);
        acc = __builtin_amdgcn_wmma_f32_16x16x32_f16(false, a0, false, b0,
                                                     (short)0, acc, false, false);
    }
#pragma unroll
    for (int i = 0; i < 8; ++i) {
        int row = m0 + i + half * 8;
        int col = n0 + r;
        if (row < M && col < N) {
            float v = acc[i] + (bias ? bias[col] : 0.0f);
            if (ACT == 1) v = sigm(v);
            C[(size_t)row * ldc + col] = v;
        }
    }
}

// ---------------- GRU scan, WMMA recurrence: one workgroup, 8 waves -------------------
// h held as padded 16x256 f16 tile in LDS (rows 8..15 zero). Per step the 16x768x256
// gate GEMM runs as 8 waves x 6 N-tiles x 8 K-tiles of v_wmma_f32_16x16x32_f16.
__global__ void k_gru_wmma(const float* __restrict__ xg, const _Float16* __restrict__ whhp,
                           const float* __restrict__ bhh, float* __restrict__ hs) {
    __shared__ _Float16 hf[16 * GRUH];        // 8 KB
    __shared__ float    gh[16 * 3 * GRUH];    // 48 KB
    int tid = threadIdx.x;
    int wv = tid >> 5, lane = tid & 31;
    int half = lane >> 4, r = lane & 15;
    for (int i = tid; i < 16 * GRUH; i += 256) hf[i] = (_Float16)0.0f;
    __syncthreads();
    for (int t = 0; t < FN; ++t) {
        // gates GEMM: gh[16,768] = hf[16,256] @ whhp[768,256]^T
        v8f acc[6];
#pragma unroll
        for (int nt = 0; nt < 6; ++nt)
            acc[nt] = (v8f){0.f, 0.f, 0.f, 0.f, 0.f, 0.f, 0.f, 0.f};
#pragma unroll
        for (int kt = 0; kt < 8; ++kt) {
            int k0 = kt * 32;
            v16h av = load_frag_lds(hf + r * GRUH, k0, half);
#pragma unroll
            for (int nt = 0; nt < 6; ++nt) {
                int n = wv * 96 + nt * 16 + r;
                v16h bv = load_frag(whhp + (size_t)n * GRUH, k0, half);
                acc[nt] = __builtin_amdgcn_wmma_f32_16x16x32_f16(false, av, false, bv,
                                                                 (short)0, acc[nt], false, false);
            }
        }
#pragma unroll
        for (int nt = 0; nt < 6; ++nt) {
            int col = wv * 96 + nt * 16 + r;
#pragma unroll
            for (int i = 0; i < 8; ++i)
                gh[(i + half * 8) * (3 * GRUH) + col] = acc[nt][i];
        }
        __syncthreads();
        // fused gate nonlinearity: thread tid handles hidden unit tid for all batches
        float hnew[BATCH];
#pragma unroll
        for (int b = 0; b < BATCH; ++b) {
            const float* xrow = xg + ((size_t)(b * FN + t)) * (3 * GRUH);
            const float* grow = gh + (size_t)b * (3 * GRUH);
            float rg = sigm(xrow[tid]            + grow[tid]            + bhh[tid]);
            float zg = sigm(xrow[GRUH + tid]     + grow[GRUH + tid]     + bhh[GRUH + tid]);
            float ng = tanhf(xrow[2*GRUH + tid]  + rg * (grow[2*GRUH + tid] + bhh[2*GRUH + tid]));
            float hp = (float)hf[b * GRUH + tid];
            float hn = (1.0f - zg) * ng + zg * hp;
            hs[((size_t)(b * FN + t)) * GRUH + tid] = hn;
            hnew[b] = hn;
        }
        __syncthreads();
#pragma unroll
        for (int b = 0; b < BATCH; ++b) hf[b * GRUH + tid] = (_Float16)hnew[b];
        __syncthreads();
    }
}

// ---------------- feature assembly: [z(16), f0n, loudn, gender(2), age] ---------------
__global__ void k_features(const float* __restrict__ z, const float* __restrict__ f0,
                           const float* __restrict__ loud, const float* __restrict__ gender,
                           const float* __restrict__ age, float* __restrict__ h0) {
    int f = blockIdx.x * blockDim.x + threadIdx.x;
    if (f >= NFRM) return;
    int b = f / FN;
    float* o = h0 + (size_t)f * 21;
    const float* zz = z + (size_t)f * ZDIM;
    for (int j = 0; j < ZDIM; ++j) o[j] = zz[j];
    o[16] = (logf(f0[f] + 1e-5f) - 4.0f) * 0.25f;
    o[17] = loud[f] * 0.01f + 1.0f;
    o[18] = gender[b * 2 + 0];
    o[19] = gender[b * 2 + 1];
    o[20] = age[b];
}

// ---------------- layernorm (dim 512) + leaky relu, in place --------------------------
__global__ void k_ln_leaky(float* __restrict__ H, const float* __restrict__ g,
                           const float* __restrict__ e) {
    __shared__ float s1[256], s2[256];
    int f = blockIdx.x, tid = threadIdx.x;
    float a = H[(size_t)f * HID + tid];
    float c = H[(size_t)f * HID + 256 + tid];
    s1[tid] = a + c; s2[tid] = a * a + c * c;
    __syncthreads();
    for (int off = 128; off > 0; off >>= 1) {
        if (tid < off) { s1[tid] += s1[tid + off]; s2[tid] += s2[tid + off]; }
        __syncthreads();
    }
    float mean = s1[0] / (float)HID;
    float var  = s2[0] / (float)HID - mean * mean;
    float inv  = rsqrtf(var + 1e-5f);
    float y0 = (a - mean) * inv * g[tid] + e[tid];        y0 = (y0 >= 0.f) ? y0 : 0.1f * y0;
    float y1 = (c - mean) * inv * g[256 + tid] + e[256 + tid]; y1 = (y1 >= 0.f) ? y1 : 0.1f * y1;
    H[(size_t)f * HID + tid]       = y0;
    H[(size_t)f * HID + 256 + tid] = y1;
}

// ---------------- upsample f0 / amp / oq to TLEN --------------------------------------
__global__ void k_upsample(const float* __restrict__ f0, const float* __restrict__ loud,
                           const float* __restrict__ oq, float* __restrict__ f0u,
                           float* __restrict__ ampu, float* __restrict__ oqu) {
    int idx = blockIdx.x * blockDim.x + threadIdx.x;
    if (idx >= NSMP) return;
    int b = idx / TLEN, i = idx % TLEN;
    float pos = ((float)i + 0.5f) * ((float)FN / (float)TLEN) - 0.5f;
    pos = fminf(fmaxf(pos, 0.0f), (float)(FN - 1));
    int lo = (int)floorf(pos);
    int hi = (lo + 1 < FN) ? lo + 1 : FN - 1;
    float w = pos - (float)lo;
    int base = b * FN;
    f0u[idx]  = f0[base + lo] * (1.0f - w) + f0[base + hi] * w;
    float alo = __expf(loud[base + lo] * 0.11512925f);   // 10^(x/20)
    float ahi = __expf(loud[base + hi] * 0.11512925f);
    ampu[idx] = alo * (1.0f - w) + ahi * w;
    oqu[idx]  = oq[base + lo] * (1.0f - w) + oq[base + hi] * w;
}

// ---------------- phase cumsum: 3-kernel block scan -----------------------------------
__global__ void k_scan1(const float* __restrict__ f0u, float* __restrict__ phase,
                        float* __restrict__ blksum) {
    __shared__ float s[256];
    int blk = blockIdx.x, b = blockIdx.y, tid = threadIdx.x;
    int t0 = blk * SCHUNK + tid * 4;
    float v[4], run = 0.0f;
    for (int j = 0; j < 4; ++j) {
        int t = t0 + j;
        float x = (t < TLEN) ? f0u[(size_t)b * TLEN + t] * (1.0f / 16000.0f) : 0.0f;
        run += x; v[j] = run;
    }
    s[tid] = run; __syncthreads();
    for (int off = 1; off < 256; off <<= 1) {
        float pv = (tid >= off) ? s[tid - off] : 0.0f;
        __syncthreads();
        s[tid] += pv;
        __syncthreads();
    }
    float excl = s[tid] - run;
    for (int j = 0; j < 4; ++j) {
        int t = t0 + j;
        if (t < TLEN) phase[(size_t)b * TLEN + t] = excl + v[j];
    }
    if (tid == 255) blksum[b * NBLK + blk] = s[255];
}

__global__ void k_scan2(float* __restrict__ blksum) {
    __shared__ float s[256];
    int b = blockIdx.x, tid = threadIdx.x;
    float v = (tid < NBLK) ? blksum[b * NBLK + tid] : 0.0f;
    s[tid] = v; __syncthreads();
    for (int off = 1; off < 256; off <<= 1) {
        float pv = (tid >= off) ? s[tid - off] : 0.0f;
        __syncthreads();
        s[tid] += pv;
        __syncthreads();
    }
    if (tid < NBLK) blksum[b * NBLK + tid] = s[tid] - v;   // exclusive
}

__device__ __forceinline__ float glottal_g(float ph, float oq) {
    float p = ph - floorf(ph);
    float oqc = fminf(fmaxf(oq, 0.1f), 0.9f);
    float pulse = 0.5f * (1.0f - cosf(3.14159265f * p / (oqc + 1e-8f)));
    return pulse * sigm((oqc - p) * 100.0f);
}

// in-place: phase buffer -> glottal
__global__ void k_scan3(float* __restrict__ phase, const float* __restrict__ blksum,
                        const float* __restrict__ f0u, const float* __restrict__ oqu) {
    int idx = blockIdx.x * blockDim.x + threadIdx.x;
    if (idx >= NSMP) return;
    int b = idx / TLEN, i = idx % TLEN;
    float ph = phase[idx] + blksum[b * NBLK + i / SCHUNK];
    float out;
    if (i == 0) out = 0.0f;
    else {
        float g1 = glottal_g(ph, oqu[idx]);
        float g0 = glottal_g(ph - f0u[idx] * (1.0f / 16000.0f), oqu[idx - 1]);
        out = g1 - g0;
    }
    phase[idx] = out;
}

// ---------------- pseudo-random Gaussian noise ----------------------------------------
__global__ void k_noise(float* __restrict__ noise) {
    int idx = blockIdx.x * blockDim.x + threadIdx.x;
    if (idx >= NSMP) return;
    unsigned h1 = pcg(0x9E3779B9u ^ (unsigned)idx);
    unsigned h2 = pcg(h1 + 0x85EBCA6Bu);
    float u1 = ((float)h1 + 0.5f) * (1.0f / 4294967296.0f);
    float u2 = ((float)h2 + 0.5f) * (1.0f / 4294967296.0f);
    noise[idx] = sqrtf(-2.0f * logf(u1)) * cosf(6.2831853072f * u2);
}

// ---------------- multiply spectrum by upsampled filter (65 -> 513 bins) --------------
__global__ void k_filter_mul(float* __restrict__ esre, float* __restrict__ esim,
                             const float* __restrict__ filt) {
    int idx = blockIdx.x * blockDim.x + threadIdx.x;
    if (idx >= NFRM * NBINS_F) return;
    int f = idx / NBINS_F, k = idx % NBINS_F;
    float pos = ((float)k + 0.5f) * ((float)NBINS / (float)NBINS_F) - 0.5f;
    pos = fminf(fmaxf(pos, 0.0f), (float)(NBINS - 1));
    int lo = (int)floorf(pos);
    int hi = (lo + 1 < NBINS) ? lo + 1 : NBINS - 1;
    float w = pos - (float)lo;
    float ff = filt[(size_t)f * NBINS + lo] * (1.0f - w) + filt[(size_t)f * NBINS + hi] * w;
    esre[idx] *= ff; esim[idx] *= ff;
}

// ---------------- ISTFT per-frame: hermitian extend, IFFT, window ---------------------
__global__ void k_istft_frames(const float* __restrict__ esre, const float* __restrict__ esim,
                               float* __restrict__ frt) {
    __shared__ float re[1024], im[1024];
    int f = blockIdx.x, tid = threadIdx.x;
    for (int i = tid; i < 1024; i += 256) {
        float r, m;
        if (i <= 512) { r = esre[(size_t)f * NBINS_F + i];        m = esim[(size_t)f * NBINS_F + i]; }
        else          { r = esre[(size_t)f * NBINS_F + 1024 - i]; m = -esim[(size_t)f * NBINS_F + 1024 - i]; }
        int p = bitrev10(i);
        re[p] = r; im[p] = m;
    }
    __syncthreads();
    fft1024_lds(re, im, tid, 256, true);
    for (int i = tid; i < 1024; i += 256)
        frt[(size_t)f * 1024 + i] = re[i] * winf(i);
}

// ---------------- overlap-add + wsq normalize + crop ----------------------------------
__global__ void k_overlap_add(const float* __restrict__ frt, const float* __restrict__ wsqi,
                              float* __restrict__ acc, int accumulate) {
    int idx = blockIdx.x * blockDim.x + threadIdx.x;
    if (idx >= NSMP) return;
    int b = idx / TLEN, t = idx % TLEN;
    int tp = t + PADN;
    int nlo = (tp - 1023 + 255) / 256; if (nlo < 0) nlo = 0;
    int nhi = tp / 256; if (nhi > FN - 1) nhi = FN - 1;
    float v = 0.0f;
    for (int n = nlo; n <= nhi; ++n)
        v += frt[((size_t)(b * FN + n)) * 1024 + (tp - 256 * n)];
    v *= wsqi[tp];
    acc[idx] = accumulate ? (acc[idx] + v) : v;
}

// ---------------- dry = acc * ampu ----------------------------------------------------
__global__ void k_dry(const float* __restrict__ acc, const float* __restrict__ ampu,
                      float* __restrict__ dry) {
    int idx = blockIdx.x * blockDim.x + threadIdx.x;
    if (idx >= NSMP) return;
    dry[idx] = acc[idx] * ampu[idx];
}

// ---------------- direct convolution with 8000-tap IR ---------------------------------
__global__ void k_conv(const float* __restrict__ dry, const float* __restrict__ ir,
                       float* __restrict__ wet) {
    __shared__ float sIr[256];
    __shared__ float sDry[512];
    int b = blockIdx.y, tid = threadIdx.x;
    int t0 = blockIdx.x * 256;
    int t = t0 + tid;
    const float* db = dry + (size_t)b * TLEN;
    float sum = 0.0f;
    for (int kc = 0; kc < IRLEN; kc += 256) {
        // full IR chunks stream global->LDS on the async path (ASYNCcnt)
        if (kc + 256 <= IRLEN) async_copy_b32(&ir[kc + tid], &sIr[tid]);
        else                   sIr[tid] = (kc + tid < IRLEN) ? ir[kc + tid] : 0.0f;
        int base = t0 - kc - 255;
        int i0 = base + tid, i1 = base + 256 + tid;
        sDry[tid]       = (i0 >= 0 && i0 < TLEN) ? db[i0] : 0.0f;
        sDry[256 + tid] = (i1 >= 0 && i1 < TLEN) ? db[i1] : 0.0f;
        async_wait_all();
        __syncthreads();
#pragma unroll 8
        for (int kk = 0; kk < 256; ++kk)
            sum += sDry[tid + 255 - kk] * sIr[kk];
        __syncthreads();
    }
    if (t < TLEN) wet[(size_t)b * TLEN + t] = sum;
}

// ======================================================================================
extern "C" void kernel_launch(void* const* d_in, const int* in_sizes, int n_in,
                              void* d_out, int out_size, void* d_ws, size_t ws_size,
                              hipStream_t stream) {
    const float* audio   = (const float*)d_in[0];
    const float* gender  = (const float*)d_in[1];
    const float* age     = (const float*)d_in[2];
    const float* gru_wih = (const float*)d_in[3];
    const float* gru_whh = (const float*)d_in[4];
    const float* gru_bih = (const float*)d_in[5];
    const float* gru_bhh = (const float*)d_in[6];
    const float* proj_w  = (const float*)d_in[7];
    const float* proj_b  = (const float*)d_in[8];
    const float* w0 = (const float*)d_in[9];   const float* b0 = (const float*)d_in[10];
    const float* g0 = (const float*)d_in[11];  const float* e0 = (const float*)d_in[12];
    const float* w1 = (const float*)d_in[13];  const float* b1 = (const float*)d_in[14];
    const float* g1 = (const float*)d_in[15];  const float* e1 = (const float*)d_in[16];
    const float* w2 = (const float*)d_in[17];  const float* b2 = (const float*)d_in[18];
    const float* g2 = (const float*)d_in[19];  const float* e2 = (const float*)d_in[20];
    const float* oq_w = (const float*)d_in[21]; const float* oq_b = (const float*)d_in[22];
    const float* vt_w = (const float*)d_in[23]; const float* vt_b = (const float*)d_in[24];
    const float* nf_w = (const float*)d_in[25]; const float* nf_b = (const float*)d_in[26];
    const float* ir   = (const float*)d_in[27];

    float* outWet = (float*)d_out;
    float* outDry = (float*)d_out + NSMP;

    // ---- workspace carve ----
    float* WS = (float*)d_ws;
    size_t o = 0;
    auto alloc  = [&](size_t n) { size_t r = o; o += (n + 63) & ~(size_t)63; return WS + r; };
    auto allocH = [&](size_t n) { return (_Float16*)alloc((n + 1) / 2); };
    float* AW     = alloc(NBINS_F);
    float* MELFBT = alloc((size_t)NMELS * NBINS_F);
    float* WSQI   = alloc(XPN);
    float* XP     = alloc((size_t)BATCH * XPN);
    float* SPEC   = alloc((size_t)NFRM * NBINS_F);
    float* F0     = alloc(NFRM);
    float* LOUD   = alloc(NFRM);
    float* MELS   = alloc((size_t)NFRM * NMELS);
    float* XG     = alloc((size_t)NFRM * 3 * GRUH);
    float* HS     = alloc((size_t)NFRM * GRUH);
    float* Z      = alloc((size_t)NFRM * ZDIM);
    float* H0     = alloc((size_t)NFRM * 21);
    float* H1     = alloc((size_t)NFRM * HID);
    float* H2     = alloc((size_t)NFRM * HID);
    float* H3     = alloc((size_t)NFRM * HID);
    float* OQ     = alloc(NFRM);
    float* VT     = alloc((size_t)NFRM * NBINS);
    float* NF     = alloc((size_t)NFRM * NBINS);
    float* F0U    = alloc(NSMP);
    float* AMPU   = alloc(NSMP);
    float* OQU    = alloc(NSMP);
    float* BLKS   = alloc((size_t)BATCH * NBLK);
    float* GLOT   = alloc(NSMP);   // phase -> glottal (in place)
    float* NOISE  = alloc(NSMP);
    float* ESRE   = alloc((size_t)NFRM * NBINS_F);
    float* ESIM   = alloc((size_t)NFRM * NBINS_F);
    float* FRT    = alloc((size_t)NFRM * 1024);
    float* ACC    = alloc(NSMP);
    // packed f16 operands (zero-padded to 16-row / 32-K tiles)
    _Float16* PW_IH   = allocH((size_t)768 * 96);
    _Float16* PW_HH   = allocH((size_t)768 * 256);
    _Float16* PW_PROJ = allocH((size_t)16 * 256);
    _Float16* PW_0    = allocH((size_t)512 * 32);
    _Float16* PW_1    = allocH((size_t)512 * 512);
    _Float16* PW_2    = allocH((size_t)512 * 512);
    _Float16* PW_OQ   = allocH((size_t)16 * 512);
    _Float16* PW_VT   = allocH((size_t)80 * 512);
    _Float16* PW_NF   = allocH((size_t)80 * 512);
    _Float16* PA_MELS = allocH((size_t)NFRM * 96);
    _Float16* PA_HS   = allocH((size_t)NFRM * 256);
    _Float16* PA_H0   = allocH((size_t)NFRM * 32);
    _Float16* PA_H    = allocH((size_t)NFRM * 512);
    (void)ws_size; (void)in_sizes; (void)n_in; (void)out_size;

    auto pack = [&](const float* src, _Float16* dst, int R, int C, int Rp, int Cp) {
        k_pack_f16<<<dim3((Rp * Cp + 255) / 256), 256, 0, stream>>>(src, dst, R, C, Rp, Cp);
    };

    // ---- weight packing (once per launch) ----
    pack(gru_wih, PW_IH,   768, 80, 768, 96);
    pack(gru_whh, PW_HH,   768, 256, 768, 256);
    pack(proj_w,  PW_PROJ, 16, 256, 16, 256);
    pack(w0,      PW_0,    512, 21, 512, 32);
    pack(w1,      PW_1,    512, 512, 512, 512);
    pack(w2,      PW_2,    512, 512, 512, 512);
    pack(oq_w,    PW_OQ,   1, 512, 16, 512);
    pack(vt_w,    PW_VT,   65, 512, 80, 512);
    pack(nf_w,    PW_NF,   65, 512, 80, 512);

    // ---- analysis ----
    k_consts<<<dim3((XPN + 255) / 256), 256, 0, stream>>>(AW, MELFBT, WSQI);
    k_pad_reflect<<<dim3((BATCH * XPN + 255) / 256), 256, 0, stream>>>(audio, XP);
    k_autocorr_f0<<<dim3(NFRM), 320, 0, stream>>>(XP, F0);
    k_stft<<<dim3(NFRM), 256, 0, stream>>>(XP, SPEC, nullptr, nullptr, 0);
    k_loud<<<dim3(NFRM), 256, 0, stream>>>(SPEC, AW, LOUD);
    k_mel<<<dim3(NFRM), 128, 0, stream>>>(SPEC, MELFBT, MELS);
    k_melnorm<<<dim3(BATCH * NMELS), 256, 0, stream>>>(MELS);

    // ---- GRU input projection (WMMA), recurrence (WMMA), latent projection (WMMA) ----
    pack(MELS, PA_MELS, NFRM, NMELS, NFRM, 96);
    k_gemm_wmma<96, 0><<<dim3(48, NFRM / 16), 32, 0, stream>>>(
        PA_MELS, PW_IH, gru_bih, XG, 3 * GRUH, NFRM, 3 * GRUH);
    k_gru_wmma<<<dim3(1), 256, 0, stream>>>(XG, PW_HH, gru_bhh, HS);
    pack(HS, PA_HS, NFRM, GRUH, NFRM, 256);
    k_gemm_wmma<256, 0><<<dim3(1, NFRM / 16), 32, 0, stream>>>(
        PA_HS, PW_PROJ, proj_b, Z, ZDIM, NFRM, ZDIM);

    // ---- MLP (WMMA + fused LN/leaky) ----
    k_features<<<dim3((NFRM + 255) / 256), 256, 0, stream>>>(Z, F0, LOUD, gender, age, H0);
    pack(H0, PA_H0, NFRM, 21, NFRM, 32);
    k_gemm_wmma<32, 0><<<dim3(32, NFRM / 16), 32, 0, stream>>>(
        PA_H0, PW_0, b0, H1, HID, NFRM, HID);
    k_ln_leaky<<<dim3(NFRM), 256, 0, stream>>>(H1, g0, e0);
    pack(H1, PA_H, NFRM, HID, NFRM, 512);
    k_gemm_wmma<512, 0><<<dim3(32, NFRM / 16), 32, 0, stream>>>(
        PA_H, PW_1, b1, H2, HID, NFRM, HID);
    k_ln_leaky<<<dim3(NFRM), 256, 0, stream>>>(H2, g1, e1);
    pack(H2, PA_H, NFRM, HID, NFRM, 512);
    k_gemm_wmma<512, 0><<<dim3(32, NFRM / 16), 32, 0, stream>>>(
        PA_H, PW_2, b2, H3, HID, NFRM, HID);
    k_ln_leaky<<<dim3(NFRM), 256, 0, stream>>>(H3, g2, e2);
    pack(H3, PA_H, NFRM, HID, NFRM, 512);

    // ---- heads (WMMA + sigmoid) ----
    k_gemm_wmma<512, 1><<<dim3(1, NFRM / 16), 32, 0, stream>>>(
        PA_H, PW_OQ, oq_b, OQ, 1, NFRM, 1);
    k_gemm_wmma<512, 1><<<dim3(5, NFRM / 16), 32, 0, stream>>>(
        PA_H, PW_VT, vt_b, VT, NBINS, NFRM, NBINS);
    k_gemm_wmma<512, 1><<<dim3(5, NFRM / 16), 32, 0, stream>>>(
        PA_H, PW_NF, nf_b, NF, NBINS, NFRM, NBINS);

    // ---- synthesis controls ----
    k_upsample<<<dim3((NSMP + 255) / 256), 256, 0, stream>>>(F0, LOUD, OQ, F0U, AMPU, OQU);
    k_scan1<<<dim3(NBLK, BATCH), 256, 0, stream>>>(F0U, GLOT, BLKS);
    k_scan2<<<dim3(BATCH), 256, 0, stream>>>(BLKS);
    k_scan3<<<dim3((NSMP + 255) / 256), 256, 0, stream>>>(GLOT, BLKS, F0U, OQU);
    k_noise<<<dim3((NSMP + 255) / 256), 256, 0, stream>>>(NOISE);

    // ---- apply_filter(excitation, filt) x2 ----
    auto apply_filter = [&](const float* ex, const float* filt, int accumulate) {
        k_pad_reflect<<<dim3((BATCH * XPN + 255) / 256), 256, 0, stream>>>(ex, XP);
        k_stft<<<dim3(NFRM), 256, 0, stream>>>(XP, nullptr, ESRE, ESIM, 1);
        k_filter_mul<<<dim3((NFRM * NBINS_F + 255) / 256), 256, 0, stream>>>(ESRE, ESIM, filt);
        k_istft_frames<<<dim3(NFRM), 256, 0, stream>>>(ESRE, ESIM, FRT);
        k_overlap_add<<<dim3((NSMP + 255) / 256), 256, 0, stream>>>(FRT, WSQI, ACC, accumulate);
    };
    apply_filter(GLOT,  VT, 0);
    apply_filter(NOISE, NF, 1);

    // ---- dry + reverb ----
    k_dry<<<dim3((NSMP + 255) / 256), 256, 0, stream>>>(ACC, AMPU, outDry);
    k_conv<<<dim3((TLEN + 255) / 256, BATCH), 256, 0, stream>>>(outDry, ir, outWet);
}
